// WeightedHAN_7172595384487
// MI455X (gfx1250) — matrix operand order repaired and checked by reading, not compiled, not run on hardware
//
#include <hip/hip_runtime.h>
#include <hip/hip_bf16.h>
#include <math.h>

#define B_ 2
#define N_ 5000
#define C_ 128
#define E_ 65536
#define T_ 4

typedef __attribute__((ext_vector_type(16))) __bf16 v16bf;
typedef __attribute__((ext_vector_type(8)))  float  v8f;

// ---------- monotone float<->uint encoding for atomic max over signed floats ----------
__device__ __forceinline__ unsigned enc_ord(float f) {
  unsigned u = __float_as_uint(f);
  return (u & 0x80000000u) ? ~u : (u | 0x80000000u);
}
__device__ __forceinline__ float dec_ord(unsigned u) {
  return __uint_as_float((u & 0x80000000u) ? (u ^ 0x80000000u) : ~u);
}

// edge type tables: EDGE_TYPES = ((0,0),(1,1),(0,1),(1,0))
__device__ __forceinline__ int src_t(int t) { return t & 1; }
__device__ __forceinline__ int dst_t(int t) { return (t & 1) ^ (t >> 1); }

// ---------------------------------------------------------------------------
// bf16-WMMA GEMM: OUT[M,128] = act( A @ W + bias )
//   A = A0 (rows x 128) for K<128, A1 for K>=128 (KTOT = 128 or 256)
//   W row-major [KTOT,128]
//   ACT: 0 = none, 1 = tanh(relu(x)),
//        2 = score-reduce: atomicAdd(score_out, sum tanh(x)*qvec[col]*scale)
// Block: 256 thr = 8 waves. Wave w: rows blockIdx.x*32 + (w>>2)*16,
//        cols (w&3)*32 .. +31  -> one A fragment feeds TWO WMMAs per k-step.
// KTOT templated so the K-loop fully unrolls (8 or 16 v_wmma per wave).
// ---------------------------------------------------------------------------
template <int KTOT, int ACT>
__global__ __launch_bounds__(256)
void gemm_wmma(const float* __restrict__ A0, const float* __restrict__ A1,
               const float* __restrict__ W,  const float* __restrict__ bias,
               float* __restrict__ OUT, int M,
               const float* __restrict__ qvec, float* __restrict__ score_out,
               float score_scale) {
  const int lane = threadIdx.x & 31;
  const int wave = threadIdx.x >> 5;
  const int row0 = blockIdx.x * 32 + (wave >> 2) * 16;
  const int col0 = (wave & 3) * 32;
  const int lm = lane & 15;   // A row in tile / B column in tile
  const int kh = lane >> 4;   // lane-half selector
  int arow = row0 + lm; if (arow >= M) arow = M - 1;  // clamp (masked on output)

  v8f acc0 = {}, acc1 = {};
#pragma unroll
  for (int k0 = 0; k0 < KTOT; k0 += 32) {
    const float* Abase = (k0 < 128) ? A0 : A1;
    const int koff = (k0 < 128) ? k0 : (k0 - 128);
    const float* ap = Abase + (size_t)arow * C_ + koff + kh * 8;
    v16bf a, b0, b1;
#pragma unroll
    for (int e = 0; e < 8; ++e) {         // A: elems 0..7 -> K=kh*8+e ; 8..15 -> +16
      a[e]     = (__bf16)ap[e];
      a[e + 8] = (__bf16)ap[16 + e];
    }
    const float* wp = W + (size_t)(k0 + kh * 8) * C_ + col0 + lm;
#pragma unroll
    for (int e = 0; e < 16; ++e) {        // B: elems 0..7 -> K=kh*8+e ; 8..15 -> +16, col = lm
      b0[e] = (__bf16)wp[(size_t)(e + (e >= 8 ? 8 : 0)) * C_];
      b1[e] = (__bf16)wp[(size_t)(e + (e >= 8 ? 8 : 0)) * C_ + 16];
    }
    acc0 = __builtin_amdgcn_wmma_f32_16x16x32_bf16(
        false, a, false, b0, (short)0, acc0, false, false);
    acc1 = __builtin_amdgcn_wmma_f32_16x16x32_bf16(
        false, a, false, b1, (short)0, acc1, false, false);
  }

  const int gcol0 = col0 + lm, gcol1 = gcol0 + 16;
  const float bv0 = bias ? bias[gcol0] : 0.0f;
  const float bv1 = bias ? bias[gcol1] : 0.0f;
  if (ACT == 2) {
    const float qv0 = qvec[gcol0] * score_scale;
    const float qv1 = qvec[gcol1] * score_scale;
    float s = 0.f;
#pragma unroll
    for (int r = 0; r < 8; ++r) {
      int grow = row0 + kh * 8 + r;
      if (grow < M) {
        s += tanhf(acc0[r] + bv0) * qv0;
        s += tanhf(acc1[r] + bv1) * qv1;
      }
    }
    for (int off = 16; off > 0; off >>= 1) s += __shfl_down(s, off, 32);
    if (lane == 0) atomicAdd(score_out, s);
  } else {
#pragma unroll
    for (int r = 0; r < 8; ++r) {
      int grow = row0 + kh * 8 + r;       // D: M = kh*8 + r, N = lm
      if (grow < M) {
        float v0 = acc0[r] + bv0;
        float v1 = acc1[r] + bv1;
        if (ACT == 1) { v0 = tanhf(fmaxf(v0, 0.f)); v1 = tanhf(fmaxf(v1, 0.f)); }
        OUT[(size_t)grow * C_ + gcol0] = v0;
        OUT[(size_t)grow * C_ + gcol1] = v1;
      }
    }
  }
}

// ---------------- zero-fill accumulation buffers ----------------
__global__ void zero_kernel(float* __restrict__ p, size_t count) {
  size_t i = (size_t)blockIdx.x * 256 + threadIdx.x;
  if (i < count) p[i] = 0.f;
}

// ---- rank-1 fold of edge-feature path:  v3 = eproj_w @ W3 ; c3 = eproj_b @ W3 + att_b
__global__ void v3c3_kernel(const float* __restrict__ att_w, const float* __restrict__ att_b,
                            const float* __restrict__ eproj_w, const float* __restrict__ eproj_b,
                            float* __restrict__ V3, float* __restrict__ C3) {
  int t = blockIdx.x, n = threadIdx.x;  // T_ blocks x 128 threads
  const float* W3 = att_w + ((size_t)t * 384 + 256) * C_ + n;
  float v = 0.f, c = 0.f;
  for (int k = 0; k < C_; ++k) {
    float w = W3[(size_t)k * C_];
    v += eproj_w[t * C_ + k] * w;
    c += eproj_b[t * C_ + k] * w;
  }
  V3[t * C_ + n] = v;
  C3[t * C_ + n] = c + att_b[t * C_ + n];
}

// ---- per-node attention scalars: a_src = <xn[s_t], lin_src>, a_dst = <xn[d_t], lin_dst>
__global__ void ascore_kernel(const float* __restrict__ XN,
                              const float* __restrict__ lin_src, const float* __restrict__ lin_dst,
                              float* __restrict__ ASRC, float* __restrict__ ADST) {
  int idx = blockIdx.x * 256 + threadIdx.x;
  if (idx >= B_ * T_ * N_) return;
  int n = idx % N_, bt = idx / N_, t = bt % T_, b = bt / T_;
  const float* xs = XN + (((size_t)(b * 2 + src_t(t))) * N_ + n) * C_;
  const float* xd = XN + (((size_t)(b * 2 + dst_t(t))) * N_ + n) * C_;
  const float* ls = lin_src + t * C_;
  const float* ld = lin_dst + t * C_;
  float as = 0.f, ad = 0.f;
  for (int c = 0; c < C_; ++c) { as += xs[c] * ls[c]; ad += xd[c] * ld[c]; }
  ASRC[idx] = as; ADST[idx] = ad;
}

// ---- edge pass 1: segment max of leaky_relu(a_src[src]+a_dst[dst]) over dst ----
__global__ void edge_max_kernel(const int* __restrict__ eidx,
                                const float* __restrict__ ASRC, const float* __restrict__ ADST,
                                unsigned* __restrict__ SMAX) {
  int e = blockIdx.x * 256 + threadIdx.x;
  int tb = blockIdx.y, b = tb & 1, t = tb >> 1;
  const int* src = eidx + (((size_t)t * B_ + b) * 2) * E_;
  int s = src[e], d = src[E_ + e];
  int base = (b * T_ + t) * N_;
  float a = ASRC[base + s] + ADST[base + d];
  a = a > 0.f ? a : 0.2f * a;
  atomicMax(&SMAX[base + d], enc_ord(a));
}

// ---- edge pass 2: denom = segment_sum exp(alpha - max) ----
__global__ void edge_den_kernel(const int* __restrict__ eidx,
                                const float* __restrict__ ASRC, const float* __restrict__ ADST,
                                const unsigned* __restrict__ SMAX, float* __restrict__ DEN) {
  int e = blockIdx.x * 256 + threadIdx.x;
  int tb = blockIdx.y, b = tb & 1, t = tb >> 1;
  const int* src = eidx + (((size_t)t * B_ + b) * 2) * E_;
  int s = src[e], d = src[E_ + e];
  int base = (b * T_ + t) * N_;
  float a = ASRC[base + s] + ADST[base + d];
  a = a > 0.f ? a : 0.2f * a;
  atomicAdd(&DEN[base + d], __expf(a - dec_ord(SMAX[base + d])));
}

// ---- edge pass 3: one wave per edge; aggr[dst] += xj * a * (Y1[src]+Y2[dst]+ew*v3+c3)
__global__ __launch_bounds__(256)
void edge_aggr_kernel(const int* __restrict__ eidx, const float* __restrict__ ew,
                      const float* __restrict__ ASRC, const float* __restrict__ ADST,
                      const unsigned* __restrict__ SMAX, const float* __restrict__ DEN,
                      const float* __restrict__ Y1, const float* __restrict__ Y2,
                      const float* __restrict__ XN,
                      const float* __restrict__ V3, const float* __restrict__ C3,
                      float* __restrict__ AGGR) {
  int lane = threadIdx.x & 31;
  int e = (blockIdx.x * 256 + threadIdx.x) >> 5;   // edge id (8 waves / block)
  int tb = blockIdx.y, b = tb & 1, t = tb >> 1;
  const int* src = eidx + (((size_t)t * B_ + b) * 2) * E_;
  int s = src[e], d = src[E_ + e];
  int base = (b * T_ + t) * N_;
  float al = ASRC[base + s] + ADST[base + d];
  al = al > 0.f ? al : 0.2f * al;
  float a = __expf(al - dec_ord(SMAX[base + d])) / DEN[base + d];
  float w = ew[((size_t)t * B_ + b) * E_ + e];

  const float4* y1 = (const float4*)(Y1 + ((size_t)base + s) * C_);
  const float4* y2 = (const float4*)(Y2 + ((size_t)base + d) * C_);
  const float4* xj = (const float4*)(XN + (((size_t)(b * 2 + src_t(t))) * N_ + s) * C_);
  const float4* v3 = (const float4*)(V3 + t * C_);
  const float4* c3 = (const float4*)(C3 + t * C_);
  float* outp = AGGR + ((size_t)base + d) * C_;

  float4 p = y1[lane], q4 = y2[lane], x4 = xj[lane], vv = v3[lane], cc = c3[lane];
  atomicAdd(&outp[lane * 4 + 0], x4.x * a * (p.x + q4.x + w * vv.x + cc.x));
  atomicAdd(&outp[lane * 4 + 1], x4.y * a * (p.y + q4.y + w * vv.y + cc.y));
  atomicAdd(&outp[lane * 4 + 2], x4.z * a * (p.z + q4.z + w * vv.z + cc.z));
  atomicAdd(&outp[lane * 4 + 3], x4.w * a * (p.w + q4.w + w * vv.w + cc.w));
}

// ---- final: 2-way semantic softmax per destination node type, write output ----
__global__ void fuse_kernel(const float* __restrict__ O, const float* __restrict__ SCORE,
                            float* __restrict__ out) {
  size_t idx = (size_t)blockIdx.x * 256 + threadIdx.x;
  if (idx >= (size_t)B_ * 2 * N_ * C_) return;
  int c = idx & (C_ - 1);
  size_t r = idx >> 7;
  int n = (int)(r % N_);
  int nt = (int)((r / N_) % 2);
  int b = (int)(r / (2 * N_));
  int g0 = nt ? 1 : 0;   // outs[0] = [et0, et3]; outs[1] = [et1, et2]
  int g1 = nt ? 2 : 3;
  float s0 = SCORE[b * T_ + g0], s1 = SCORE[b * T_ + g1];
  float m = fmaxf(s0, s1);
  float w0 = __expf(s0 - m), w1 = __expf(s1 - m);
  float inv = 1.f / (w0 + w1);
  float o0 = O[(((size_t)(b * T_ + g0)) * N_ + n) * C_ + c];
  float o1 = O[(((size_t)(b * T_ + g1)) * N_ + n) * C_ + c];
  out[idx] = (w0 * o0 + w1 * o1) * inv;
}

extern "C" void kernel_launch(void* const* d_in, const int* in_sizes, int n_in,
                              void* d_out, int out_size, void* d_ws, size_t ws_size,
                              hipStream_t stream) {
  (void)in_sizes; (void)n_in; (void)out_size; (void)ws_size;
  const float* x       = (const float*)d_in[0];
  const int*   eidx    = (const int*)  d_in[1];
  const float* ew      = (const float*)d_in[2];
  const float* proj_w  = (const float*)d_in[3];
  const float* proj_b  = (const float*)d_in[4];
  const float* q       = (const float*)d_in[5];
  const float* klin_w  = (const float*)d_in[6];
  const float* klin_b  = (const float*)d_in[7];
  const float* lin_src = (const float*)d_in[8];
  const float* lin_dst = (const float*)d_in[9];
  const float* eproj_w = (const float*)d_in[10];
  const float* eproj_b = (const float*)d_in[11];
  const float* agg_w   = (const float*)d_in[12];
  const float* agg_b   = (const float*)d_in[13];
  const float* att_w   = (const float*)d_in[14];
  const float* att_b   = (const float*)d_in[15];
  float* out = (float*)d_out;
  float* ws  = (float*)d_ws;

  // workspace layout (floats)
  size_t o = 0;
  float* XN   = ws + o; o += (size_t)B_ * 2 * N_ * C_;   // projected nodes [B][2][N][C]
  float* Y1   = ws + o; o += (size_t)B_ * T_ * N_ * C_;  // xn[s_t] @ W1
  float* Y2   = ws + o; o += (size_t)B_ * T_ * N_ * C_;  // xn[d_t] @ W2
  float* Ovt  = ws + o; o += (size_t)B_ * T_ * N_ * C_;  // tanh(relu(agg))
  float* ASRC = ws + o; o += (size_t)B_ * T_ * N_;
  float* ADST = ws + o; o += (size_t)B_ * T_ * N_;
  float* V3   = ws + o; o += (size_t)T_ * C_;
  float* C3   = ws + o; o += (size_t)T_ * C_;
  float* ZS   = ws + o;                                  // zeroed region starts here
  unsigned* SMAX = (unsigned*)(ws + o); o += (size_t)B_ * T_ * N_;
  float* DEN  = ws + o; o += (size_t)B_ * T_ * N_;
  float* AGGR = ws + o; o += (size_t)B_ * T_ * N_ * C_;
  float* SCORE= ws + o; o += (size_t)B_ * T_;
  size_t zcount = (size_t)((ws + o) - ZS);

  zero_kernel<<<(unsigned)((zcount + 255) / 256), 256, 0, stream>>>(ZS, zcount);

  const dim3 gG((N_ + 31) / 32);   // 32 output rows per block
  // 1) node-type projections: xn[b][nt] = x_part @ proj_w[nt] + proj_b[nt]
  for (int b = 0; b < B_; ++b)
    for (int nt = 0; nt < 2; ++nt)
      gemm_wmma<128, 0><<<gG, 256, 0, stream>>>(
          x + ((size_t)b * 2 * N_ + (size_t)nt * N_) * C_, nullptr,
          proj_w + (size_t)nt * C_ * C_, proj_b + nt * C_,
          XN + ((size_t)(b * 2 + nt)) * N_ * C_, N_, nullptr, nullptr, 0.f);

  // 2) rank-1 edge-feature fold + per-node attention scalars
  v3c3_kernel<<<T_, C_, 0, stream>>>(att_w, att_b, eproj_w, eproj_b, V3, C3);
  ascore_kernel<<<(B_ * T_ * N_ + 255) / 256, 256, 0, stream>>>(XN, lin_src, lin_dst, ASRC, ADST);

  // 3) node-level halves of the per-edge attention GEMM
  for (int b = 0; b < B_; ++b)
    for (int t = 0; t < T_; ++t) {
      int st = t & 1, dt = st ^ (t >> 1);
      gemm_wmma<128, 0><<<gG, 256, 0, stream>>>(
          XN + ((size_t)(b * 2 + st)) * N_ * C_, nullptr,
          att_w + (size_t)t * 384 * C_, nullptr,
          Y1 + ((size_t)(b * T_ + t)) * N_ * C_, N_, nullptr, nullptr, 0.f);
      gemm_wmma<128, 0><<<gG, 256, 0, stream>>>(
          XN + ((size_t)(b * 2 + dt)) * N_ * C_, nullptr,
          att_w + (size_t)t * 384 * C_ + (size_t)C_ * C_, nullptr,
          Y2 + ((size_t)(b * T_ + t)) * N_ * C_, N_, nullptr, nullptr, 0.f);
    }

  // 4) edge passes: segment max, exp-sum, weighted aggregation
  const dim3 gE(E_ / 256, T_ * B_);
  edge_max_kernel<<<gE, 256, 0, stream>>>(eidx, ASRC, ADST, SMAX);
  edge_den_kernel<<<gE, 256, 0, stream>>>(eidx, ASRC, ADST, SMAX, DEN);
  const dim3 gA((E_ * 32) / 256, T_ * B_);
  edge_aggr_kernel<<<gA, 256, 0, stream>>>(eidx, ew, ASRC, ADST, SMAX, DEN,
                                           Y1, Y2, XN, V3, C3, AGGR);

  // 5) update GEMM: o = tanh(relu(concat([aggr, xdst]) @ agg_w + agg_b))
  for (int b = 0; b < B_; ++b)
    for (int t = 0; t < T_; ++t) {
      int dt = (t & 1) ^ (t >> 1);
      gemm_wmma<256, 1><<<gG, 256, 0, stream>>>(
          AGGR + ((size_t)(b * T_ + t)) * N_ * C_,
          XN + ((size_t)(b * 2 + dt)) * N_ * C_,
          agg_w + (size_t)t * 2 * C_ * C_, agg_b + t * C_,
          Ovt + ((size_t)(b * T_ + t)) * N_ * C_, N_, nullptr, nullptr, 0.f);
    }

  // 6) semantic-attention scores: sum_c q[c] * mean_n tanh(o @ klin_w + klin_b)
  for (int b = 0; b < B_; ++b)
    for (int t = 0; t < T_; ++t) {
      int nt = (t & 1) ^ (t >> 1);
      gemm_wmma<128, 2><<<gG, 256, 0, stream>>>(
          Ovt + ((size_t)(b * T_ + t)) * N_ * C_, nullptr,
          klin_w + (size_t)nt * C_ * C_, klin_b + nt * C_,
          nullptr, N_, q + nt * C_, SCORE + b * T_ + t, 1.0f / (float)N_);
    }

  // 7) softmax-weighted fusion into output [B, 2N, C]
  fuse_kernel<<<(unsigned)(((size_t)B_ * 2 * N_ * C_ + 255) / 256), 256, 0, stream>>>(
      Ovt, SCORE, out);
}
